// DepthSeparableConv2d_conv4_1_37572373905423
// MI455X (gfx1250) — matrix operand order repaired and checked by reading, not compiled, not run on hardware
//
#include <hip/hip_runtime.h>
#include <hip/hip_bf16.h>

// ---- problem constants (match reference) ----
constexpr int kB    = 64;
constexpr int kCIN  = 256;
constexpr int kCOUT = 512;
constexpr int kH    = 56;
constexpr int kW    = 56;
constexpr int kHW   = kH * kW;          // 3136 = 64 * 49
constexpr float kEPS = 1e-5f;
constexpr float kDW_THRESH = 4.0f;
constexpr float kPW_THRESH = 1e-3f;

typedef __attribute__((ext_vector_type(16))) __bf16 v16bf;
typedef __attribute__((ext_vector_type(8)))  __bf16 v8bf;
typedef __attribute__((ext_vector_type(8)))  float  v8f;

// ---------------------------------------------------------------------------
// Kernel 1: fold pointwise BN scale into bf16 weights + fused bias; zero maxbuf
// grid: COUT blocks of 256 threads (thread == input channel)
// ---------------------------------------------------------------------------
__global__ __launch_bounds__(256) void prep_kernel(
    const float* __restrict__ pw_w, const float* __restrict__ pw_b,
    const float* __restrict__ gamma, const float* __restrict__ beta,
    const float* __restrict__ mean, const float* __restrict__ var,
    __hip_bfloat16* __restrict__ wA, float* __restrict__ biasAdj,
    unsigned* __restrict__ maxbuf)
{
    const int o = blockIdx.x;
    const int t = threadIdx.x;
    const float inv = gamma[o] * rsqrtf(var[o] + kEPS);
    wA[(size_t)o * kCIN + t] = __float2bfloat16(pw_w[(size_t)o * kCIN + t] * inv);
    if (t == 0)
        biasAdj[o] = pw_b[o] * inv + beta[o] - mean[o] * inv;
    // maxbuf has kB*kCOUT = 512*64 entries; block o zeros its 64-entry stripe
    if (t < kB)
        maxbuf[(size_t)o * kB + t] = 0u;
}

// ---------------------------------------------------------------------------
// Kernel 2: depthwise 3x3 + bias + BN + ReLU + in-block slab cut -> bf16 y
// One block per (b, c) slab. y layout: [b][c][hw] (coalesced stores).
// ---------------------------------------------------------------------------
__global__ __launch_bounds__(256) void dw_kernel(
    const float* __restrict__ x, const float* __restrict__ dw_w,
    const float* __restrict__ dw_b, const float* __restrict__ gamma,
    const float* __restrict__ beta, const float* __restrict__ mean,
    const float* __restrict__ var, __hip_bfloat16* __restrict__ y)
{
    __shared__ float tile[58 * 58];
    __shared__ float red[256];

    const int bc  = blockIdx.x;            // b*kCIN + c
    const int c   = bc & (kCIN - 1);
    const int tid = threadIdx.x;

    // stage 56x56 slab with zero halo -> 58x58
    const float* xs = x + (size_t)bc * kHW;
    for (int i = tid; i < 58 * 58; i += 256) {
        const int th = i / 58 - 1;
        const int tw = i % 58 - 1;
        float v = 0.0f;
        if ((unsigned)th < (unsigned)kH && (unsigned)tw < (unsigned)kW)
            v = xs[th * kW + tw];
        tile[i] = v;
    }

    const float w0 = dw_w[c * 9 + 0], w1 = dw_w[c * 9 + 1], w2 = dw_w[c * 9 + 2];
    const float w3 = dw_w[c * 9 + 3], w4 = dw_w[c * 9 + 4], w5 = dw_w[c * 9 + 5];
    const float w6 = dw_w[c * 9 + 6], w7 = dw_w[c * 9 + 7], w8 = dw_w[c * 9 + 8];
    const float inv = gamma[c] * rsqrtf(var[c] + kEPS);
    const float off = (beta[c] - mean[c] * inv) + dw_b[c] * inv;

    __syncthreads();

    float vals[13];
    float lmax = 0.0f;
    #pragma unroll
    for (int k = 0; k < 13; ++k) {
        const int i = tid + k * 256;
        float v = 0.0f;
        if (i < kHW) {
            const int h = i / kW;
            const int w = i % kW;
            const float* t = &tile[h * 58 + w];   // window corner (h-1, w-1)
            float acc = t[0]   * w0 + t[1]   * w1 + t[2]   * w2
                      + t[58]  * w3 + t[59]  * w4 + t[60]  * w5
                      + t[116] * w6 + t[117] * w7 + t[118] * w8;
            v = acc * inv + off;
            v = v > 0.0f ? v : 0.0f;
        }
        vals[k] = v;
        lmax = fmaxf(lmax, v);
    }

    // block reduce max (values are >= 0 post-ReLU)
    red[tid] = lmax;
    __syncthreads();
    for (int s = 128; s > 0; s >>= 1) {
        if (tid < s) red[tid] = fmaxf(red[tid], red[tid + s]);
        __syncthreads();
    }
    const float scale = (red[0] < kDW_THRESH) ? 0.0f : 1.0f;

    __hip_bfloat16* ys = y + (size_t)bc * kHW;
    #pragma unroll
    for (int k = 0; k < 13; ++k) {
        const int i = tid + k * 256;
        if (i < kHW) ys[i] = __float2bfloat16(vals[k] * scale);
    }
}

// ---------------------------------------------------------------------------
// Kernel 3: pointwise GEMM with v_wmma_f32_16x16x32_bf16.
//   z[b][o][hw] = relu(sum_c wA[o][c]*y[b][c][hw] + biasAdj[o])
// Block tile: 128(M) x 64(N), 8 waves, wave tile 32x32 (2x2 WMMA frags),
// K loop = 8 steps of 32. B(y) tile staged via LDS transpose, A from L2.
// grid: (49 N-tiles, 4 M-tiles, 64 batches)
// ---------------------------------------------------------------------------
__global__ __launch_bounds__(256) void pw_gemm_kernel(
    const __hip_bfloat16* __restrict__ y_raw,
    const __hip_bfloat16* __restrict__ wA_raw,
    const float* __restrict__ biasAdj,
    float* __restrict__ out, unsigned* __restrict__ maxbuf)
{
    constexpr int LDSROW = 40;               // padded row stride (elements)
    __shared__ __bf16 ldsB[64 * LDSROW];     // [hw_local][c_local], padded

    const int b      = blockIdx.z;
    const int hwBase = blockIdx.x * 64;
    const int mBase  = blockIdx.y * 128;

    const int tid   = threadIdx.x;
    const int lane  = tid & 31;
    const int wave  = tid >> 5;
    const int waveM = wave & 3;      // 0..3 -> M sub-tile of 32
    const int waveN = wave >> 2;     // 0..1 -> N sub-tile of 32
    const int g     = lane >> 4;     // 16-lane group
    const int ln    = lane & 15;

    const __bf16* yb = (const __bf16*)y_raw + (size_t)b * kCIN * kHW;
    const __bf16* wA = (const __bf16*)wA_raw;

    v8f acc[2][2] = {};

    // staging assignment: thread -> (c row, 8-wide hw chunk)
    const int sr = tid >> 3;           // 0..31  (c_local)
    const int sc = (tid & 7) * 8;      // 0..56  (hw_local chunk)

    for (int kk = 0; kk < kCIN; kk += 32) {
        __syncthreads();
        // coalesced 16B global load of y row segment, transpose into LDS
        const __bf16* src = yb + (size_t)(kk + sr) * kHW + hwBase + sc;
        v8bf row = *(const v8bf*)src;
        #pragma unroll
        for (int e = 0; e < 8; ++e)
            ldsB[(sc + e) * LDSROW + sr] = row[e];
        __syncthreads();

        // A fragments: lane = row M (o), halves = K {8g..8g+7, 8g+16..8g+23}
        v16bf afrag[2];
        #pragma unroll
        for (int mi = 0; mi < 2; ++mi) {
            const __bf16* ap = wA + (size_t)(mBase + waveM * 32 + mi * 16 + ln) * kCIN
                                  + kk + 8 * g;
            *(v8bf*)&afrag[mi]       = *(const v8bf*)ap;
            *((v8bf*)&afrag[mi] + 1) = *(const v8bf*)(ap + 16);
        }
        // B fragments: lane = col N (hw), halves = K {8g..8g+7, 8g+16..8g+23}
        v16bf bfrag[2];
        #pragma unroll
        for (int ni = 0; ni < 2; ++ni) {
            const __bf16* bp = &ldsB[(waveN * 32 + ni * 16 + ln) * LDSROW + 8 * g];
            *(v8bf*)&bfrag[ni]       = *(const v8bf*)bp;
            *((v8bf*)&bfrag[ni] + 1) = *(const v8bf*)(bp + 16);
        }

        #pragma unroll
        for (int mi = 0; mi < 2; ++mi)
            #pragma unroll
            for (int ni = 0; ni < 2; ++ni)
                acc[mi][ni] = __builtin_amdgcn_wmma_f32_16x16x32_bf16(
                    false, afrag[mi], false, bfrag[ni],
                    (short)0, acc[mi][ni], false, false);
    }

    // epilogue: bias + ReLU, coalesced f32 stores, per-(b,o) max via atomics
    float* ob = out + (size_t)b * kCOUT * kHW;
    unsigned* mb = maxbuf + (size_t)b * kCOUT;

    #pragma unroll
    for (int mi = 0; mi < 2; ++mi) {
        #pragma unroll
        for (int r = 0; r < 8; ++r) {
            const int o = mBase + waveM * 32 + mi * 16 + r + 8 * g;  // C/D: M = r + 8g
            const float bias = biasAdj[o];
            float m2 = 0.0f;
            #pragma unroll
            for (int ni = 0; ni < 2; ++ni) {
                float v = acc[mi][ni][r] + bias;
                v = v > 0.0f ? v : 0.0f;
                const int hw = hwBase + waveN * 32 + ni * 16 + ln;
                ob[(size_t)o * kHW + hw] = v;
                m2 = fmaxf(m2, v);
            }
            // reduce across the 16-lane N group (stays within each half-wave)
            #pragma unroll
            for (int s = 1; s < 16; s <<= 1)
                m2 = fmaxf(m2, __shfl_xor(m2, s, 32));
            if (ln == 0)
                atomicMax(&mb[o], __float_as_uint(m2));  // vals >= 0: uint order == float order
        }
    }
}

// ---------------------------------------------------------------------------
// Kernel 4: pointwise channel cut — zero slabs whose max < 1e-3
// grid: kB*kCOUT blocks (one per slab); most return immediately
// ---------------------------------------------------------------------------
__global__ __launch_bounds__(256) void cut_kernel(
    float* __restrict__ out, const unsigned* __restrict__ maxbuf)
{
    const int slab = blockIdx.x;             // b*kCOUT + o
    const float mx = __uint_as_float(maxbuf[slab]);
    if (mx >= kPW_THRESH) return;
    float* p = out + (size_t)slab * kHW;
    for (int i = threadIdx.x; i < kHW; i += 256) p[i] = 0.0f;
}

// ---------------------------------------------------------------------------
extern "C" void kernel_launch(void* const* d_in, const int* in_sizes, int n_in,
                              void* d_out, int out_size, void* d_ws, size_t ws_size,
                              hipStream_t stream)
{
    (void)in_sizes; (void)n_in; (void)out_size; (void)ws_size;

    const float* x        = (const float*)d_in[0];
    const float* dw_w     = (const float*)d_in[1];
    const float* dw_b     = (const float*)d_in[2];
    const float* dw_gamma = (const float*)d_in[3];
    const float* dw_beta  = (const float*)d_in[4];
    const float* dw_mean  = (const float*)d_in[5];
    const float* dw_var   = (const float*)d_in[6];
    const float* pw_w     = (const float*)d_in[7];
    const float* pw_b     = (const float*)d_in[8];
    const float* pw_gamma = (const float*)d_in[9];
    const float* pw_beta  = (const float*)d_in[10];
    const float* pw_mean  = (const float*)d_in[11];
    const float* pw_var   = (const float*)d_in[12];
    float* out = (float*)d_out;

    // workspace carve-up (256B aligned)
    char* ws = (char*)d_ws;
    size_t off = 0;
    __hip_bfloat16* y  = (__hip_bfloat16*)(ws + off);
    off += (size_t)kB * kCIN * kHW * sizeof(__hip_bfloat16);   // 102.8 MB
    off = (off + 255) & ~(size_t)255;
    __hip_bfloat16* wA = (__hip_bfloat16*)(ws + off);
    off += (size_t)kCOUT * kCIN * sizeof(__hip_bfloat16);      // 256 KB
    off = (off + 255) & ~(size_t)255;
    float* biasAdj = (float*)(ws + off);
    off += (size_t)kCOUT * sizeof(float);
    off = (off + 255) & ~(size_t)255;
    unsigned* maxbuf = (unsigned*)(ws + off);                  // kB*kCOUT entries

    prep_kernel<<<kCOUT, 256, 0, stream>>>(pw_w, pw_b, pw_gamma, pw_beta,
                                           pw_mean, pw_var, wA, biasAdj, maxbuf);

    dw_kernel<<<kB * kCIN, 256, 0, stream>>>(x, dw_w, dw_b, dw_gamma, dw_beta,
                                             dw_mean, dw_var, y);

    dim3 ggrid(kHW / 64, kCOUT / 128, kB);   // (49, 4, 64)
    pw_gemm_kernel<<<ggrid, 256, 0, stream>>>(y, wA, biasAdj, out, maxbuf);

    cut_kernel<<<kB * kCOUT, 256, 0, stream>>>(out, maxbuf);
}
